// EncoderLayer_64467459113309
// MI455X (gfx1250) — matrix operand order repaired and checked by reading, not compiled, not run on hardware
//
#include <hip/hip_runtime.h>
#include <hip/hip_bf16.h>

#define D_MODEL 768
#define NUM_HEADS 12
#define HEAD_DIM 64
#define D_FF 3072
#define SEQ 2048
#define BATCH 4
#define NTOK (BATCH * SEQ) // 8192

typedef __bf16 bf16_t;
typedef __attribute__((ext_vector_type(16))) __bf16 bf16x16;
typedef __attribute__((ext_vector_type(8))) float f32x8;
typedef int v4i_gvec __attribute__((vector_size(16)));

#if defined(__has_builtin)
#if __has_builtin(__builtin_amdgcn_global_load_async_to_lds_b128)
#define HAVE_ASYNC_LDS 1
#else
#define HAVE_ASYNC_LDS 0
#endif
#if __has_builtin(__builtin_amdgcn_s_wait_asynccnt)
#define HAVE_WAIT_ASYNC 1
#else
#define HAVE_WAIT_ASYNC 0
#endif
#else
#define HAVE_ASYNC_LDS 0
#define HAVE_WAIT_ASYNC 0
#endif

union FragU {
  bf16x16 v;
  uint4 u[2];
};

// A-matrix 16x32 bf16 fragment: lane half h, row r. VGPR0-3 = k[8h..8h+7],
// VGPR4-7 = k[8h+16..8h+23]. Two 16B chunks at +0 and +16 elements.
__device__ __forceinline__ bf16x16 load_a_frag(const bf16_t* p) {
  FragU f;
  f.u[0] = *(const uint4*)(p);
  f.u[1] = *(const uint4*)(p + 16);
  return f.v;
}

// B-matrix 32x16 bf16 fragment: lane half h holds k[8h..8h+7], k[8h+16..8h+23].
__device__ __forceinline__ bf16x16 load_b_frag(const bf16_t* p) {
  FragU f;
  f.u[0] = *(const uint4*)(p);
  f.u[1] = *(const uint4*)(p + 16);
  return f.v;
}

// 16B global -> LDS staging; async path (ASYNCcnt-tracked) when available.
__device__ __forceinline__ void stage16(const bf16_t* g, bf16_t* l) {
#if HAVE_ASYNC_LDS
  __attribute__((address_space(1))) v4i_gvec* gp =
      (__attribute__((address_space(1))) v4i_gvec*)(unsigned long long)(size_t)g;
  __attribute__((address_space(3))) v4i_gvec* lp =
      (__attribute__((address_space(3))) v4i_gvec*)(unsigned int)(size_t)l;
  __builtin_amdgcn_global_load_async_to_lds_b128(gp, lp, 0, 0);
#else
  *(uint4*)l = *(const uint4*)g;
#endif
}

__device__ __forceinline__ void async_fence() {
#if HAVE_ASYNC_LDS
#if HAVE_WAIT_ASYNC
  __builtin_amdgcn_s_wait_asynccnt(0);
#else
  asm volatile("s_wait_asynccnt 0" ::: "memory");
#endif
#endif
}

// VALU-only 16-lane row reductions via DPP row_ror (no ds_bpermute traffic).
template <int CTRL>
__device__ __forceinline__ float dpp_ror(float x) {
  return __int_as_float(
      __builtin_amdgcn_update_dpp(0, __float_as_int(x), CTRL, 0xf, 0xf, true));
}
__device__ __forceinline__ float rowmax16(float x) {
  x = fmaxf(x, dpp_ror<0x121>(x));
  x = fmaxf(x, dpp_ror<0x122>(x));
  x = fmaxf(x, dpp_ror<0x124>(x));
  x = fmaxf(x, dpp_ror<0x128>(x));
  return x;
}
__device__ __forceinline__ float rowsum16(float x) {
  x += dpp_ror<0x121>(x);
  x += dpp_ror<0x122>(x);
  x += dpp_ror<0x124>(x);
  x += dpp_ror<0x128>(x);
  return x;
}

// ---------------------------------------------------------------------------
// f32 -> bf16 bulk convert (vector of 4)
// ---------------------------------------------------------------------------
__global__ __launch_bounds__(256) void cvt_kernel(const float* __restrict__ in,
                                                  bf16_t* __restrict__ out, int n4) {
  int i = blockIdx.x * 256 + threadIdx.x;
  if (i >= n4) return;
  float4 f = ((const float4*)in)[i];
  union { bf16_t b[4]; uint2 u; } o;
  o.b[0] = (bf16_t)f.x;
  o.b[1] = (bf16_t)f.y;
  o.b[2] = (bf16_t)f.z;
  o.b[3] = (bf16_t)f.w;
  ((uint2*)out)[i] = o.u;
}

// ---------------------------------------------------------------------------
// bf16 GEMM: out[M,N] = (A[M,K] @ W[N,K]^T) * oscale (+bias) (+relu)
// block = (32,8): 8 waves, 4 along M x 2 along N. Wave tile 32x64.
// OUT_MODE 0: dense [M,N].  OUT_MODE 1: scatter to Vt[b,h,e,s] (M=NTOK,N=768).
// ---------------------------------------------------------------------------
template <bool RELU, bool HAS_BIAS, int OUT_MODE, typename OutT>
__global__ __launch_bounds__(256) void gemm_kernel(const bf16_t* __restrict__ A,
                                                   const bf16_t* __restrict__ W,
                                                   const float* __restrict__ bias,
                                                   OutT* __restrict__ out, int M, int N,
                                                   int K, float oscale) {
  const int lane = threadIdx.x;
  const int wave = threadIdx.y;
  const int half = lane >> 4, r = lane & 15;
  const int m0 = blockIdx.y * 128 + (wave & 3) * 32;
  const int n0 = blockIdx.x * 128 + (wave >> 2) * 64;

  const bf16_t* Ap = A + (size_t)(m0 + r) * K + 8 * half;
  const bf16_t* Wp = W + (size_t)(n0 + r) * K + 8 * half;

  f32x8 acc[2][4] = {};

#pragma unroll 2
  for (int k = 0; k < K; k += 32) {
    bf16x16 a[2], b[4];
#pragma unroll
    for (int mt = 0; mt < 2; ++mt) a[mt] = load_a_frag(Ap + (size_t)mt * 16 * K + k);
#pragma unroll
    for (int nt = 0; nt < 4; ++nt) b[nt] = load_b_frag(Wp + (size_t)nt * 16 * K + k);
    __builtin_prefetch((const void*)(Ap + k + 256), 0, 0);
#pragma unroll
    for (int mt = 0; mt < 2; ++mt)
#pragma unroll
      for (int nt = 0; nt < 4; ++nt)
        acc[mt][nt] = __builtin_amdgcn_wmma_f32_16x16x32_bf16(
            false, a[mt], false, b[nt], (short)0, acc[mt][nt], false, false);
  }

#pragma unroll
  for (int nt = 0; nt < 4; ++nt) {
    const int col = n0 + nt * 16 + r;
    const float bv = HAS_BIAS ? bias[col] : 0.0f;
#pragma unroll
    for (int mt = 0; mt < 2; ++mt) {
#pragma unroll
      for (int v = 0; v < 8; ++v) {
        const int row = m0 + mt * 16 + half * 8 + v;
        float x = acc[mt][nt][v] * oscale + bv;
        if (RELU) x = fmaxf(x, 0.0f);
        if constexpr (OUT_MODE == 1) {
          const int bb = row >> 11, s = row & (SEQ - 1);
          const int hh = col >> 6, e = col & 63;
          out[(((size_t)(bb * NUM_HEADS + hh) * HEAD_DIM + e) << 11) + s] = (OutT)x;
        } else {
          out[(size_t)row * N + col] = (OutT)x;
        }
      }
    }
  }
}

// ---------------------------------------------------------------------------
// Flash attention (base-2 softmax; 1/sqrt(64)*log2(e) pre-folded into Q).
// grid (SEQ/128, B*H), block (32,8). Each wave: 16 Q rows.
// K/V chunks staged cooperatively into LDS (async path when available).
// ---------------------------------------------------------------------------
__global__ __launch_bounds__(256) void attn_kernel(const bf16_t* __restrict__ Q,
                                                   const bf16_t* __restrict__ Km,
                                                   const bf16_t* __restrict__ Vt,
                                                   bf16_t* __restrict__ Ctx) {
  const int lane = threadIdx.x;
  const int wave = threadIdx.y;
  const int tid = wave * 32 + lane;
  const int half = lane >> 4, r = lane & 15;
  const int bh = blockIdx.y;
  const int b = bh / NUM_HEADS, h = bh % NUM_HEADS;
  const int q0 = blockIdx.x * 128 + wave * 16;
  const size_t qrow = (size_t)(b * SEQ + q0 + r) * D_MODEL + (size_t)h * HEAD_DIM;

  __shared__ bf16_t ksh[32][64];     // key chunk:  32 keys x 64 dims (4KB)
  __shared__ bf16_t vsh[64][32];     // value chunk: 64 dims x 32 keys (4KB)
  __shared__ bf16_t psh[8][16][32];  // per-wave P tile (8KB)

  bf16x16 aq[2];
  aq[0] = load_a_frag(Q + qrow + 8 * half);
  aq[1] = load_a_frag(Q + qrow + 32 + 8 * half);

  f32x8 ctx[4] = {};
  float mrun[8], lrun[8];
#pragma unroll
  for (int v = 0; v < 8; ++v) {
    mrun[v] = -3.0e30f;
    lrun[v] = 0.0f;
  }

  // staging coordinates (one 16B chunk per thread per tensor)
  const int krr = tid >> 3, kcc = (tid & 7) * 8;
  const int vee = tid >> 2, vcc = (tid & 3) * 8;

  for (int t0 = 0; t0 < SEQ; t0 += 32) {
    __syncthreads(); // previous chunk fully consumed before overwrite
    stage16(Km + (size_t)(b * SEQ + t0 + krr) * D_MODEL + h * HEAD_DIM + kcc,
            &ksh[krr][kcc]);
    stage16(Vt + (((size_t)bh * HEAD_DIM + vee) << 11) + t0 + vcc, &vsh[vee][vcc]);
    async_fence();
    __syncthreads();

    f32x8 s[2] = {};
#pragma unroll
    for (int ks = 0; ks < 2; ++ks)
#pragma unroll
      for (int j = 0; j < 2; ++j)
        s[j] = __builtin_amdgcn_wmma_f32_16x16x32_bf16(
            false, aq[ks], false, load_b_frag(&ksh[16 * j + r][ks * 32 + 8 * half]),
            (short)0, s[j], false, false);

    // online base-2 softmax: DPP row reductions over the 16 lanes of each row
#pragma unroll
    for (int v = 0; v < 8; ++v) {
      float s0 = s[0][v];
      float s1 = s[1][v];
      float mnew = fmaxf(mrun[v], rowmax16(fmaxf(s0, s1)));
      float corr = exp2f(mrun[v] - mnew);
      mrun[v] = mnew;
      float p0 = exp2f(s0 - mnew);
      float p1 = exp2f(s1 - mnew);
      psh[wave][half * 8 + v][r] = (bf16_t)p0;
      psh[wave][half * 8 + v][16 + r] = (bf16_t)p1;
      lrun[v] = lrun[v] * corr + rowsum16(p0 + p1);
#pragma unroll
      for (int jn = 0; jn < 4; ++jn) ctx[jn][v] *= corr;
    }
    asm volatile("" ::: "memory"); // keep LDS P stores before frag reload
    // P (C-layout) -> A-layout via wave-private LDS slice; DS ops are in-order
    bf16x16 pf = load_a_frag(&psh[wave][r][8 * half]);
#pragma unroll
    for (int jn = 0; jn < 4; ++jn)
      ctx[jn] = __builtin_amdgcn_wmma_f32_16x16x32_bf16(
          false, pf, false, load_b_frag(&vsh[jn * 16 + r][8 * half]), (short)0,
          ctx[jn], false, false);
  }

#pragma unroll
  for (int v = 0; v < 8; ++v) {
    const float inv = 1.0f / lrun[v];
    const size_t row =
        (size_t)(b * SEQ + q0 + half * 8 + v) * D_MODEL + (size_t)h * HEAD_DIM;
#pragma unroll
    for (int jn = 0; jn < 4; ++jn) Ctx[row + jn * 16 + r] = (bf16_t)(ctx[jn][v] * inv);
  }
}

// ---------------------------------------------------------------------------
// out = LayerNorm(a + bres) * g + be ; optionally also bf16 copy
// one block (256 threads) per token row; D_MODEL = 768 = 3 * 256
// ---------------------------------------------------------------------------
template <bool WRITE_BF16>
__global__ __launch_bounds__(256) void add_ln_kernel(
    const float* __restrict__ a, const float* __restrict__ bres,
    const float* __restrict__ g, const float* __restrict__ be,
    float* __restrict__ outf, bf16_t* __restrict__ outb) {
  const int row = blockIdx.x;
  const int tid = threadIdx.x;
  const float* ar = a + (size_t)row * D_MODEL;
  const float* br = bres + (size_t)row * D_MODEL;
  float val[3];
  float s = 0.f, s2 = 0.f;
#pragma unroll
  for (int j = 0; j < 3; ++j) {
    int c = tid + 256 * j;
    float t = ar[c] + br[c];
    val[j] = t;
    s += t;
    s2 += t * t;
  }
#pragma unroll
  for (int m = 16; m; m >>= 1) {
    s += __shfl_xor(s, m, 32);
    s2 += __shfl_xor(s2, m, 32);
  }
  __shared__ float rs[8], rq[8];
  const int wv = tid >> 5, ln = tid & 31;
  if (ln == 0) {
    rs[wv] = s;
    rq[wv] = s2;
  }
  __syncthreads();
  float st = 0.f, sq = 0.f;
#pragma unroll
  for (int i = 0; i < 8; ++i) {
    st += rs[i];
    sq += rq[i];
  }
  const float mu = st * (1.0f / D_MODEL);
  const float var = sq * (1.0f / D_MODEL) - mu * mu;
  const float rstd = rsqrtf(var + 1e-5f);
#pragma unroll
  for (int j = 0; j < 3; ++j) {
    int c = tid + 256 * j;
    float o = (val[j] - mu) * rstd * g[c] + be[c];
    outf[(size_t)row * D_MODEL + c] = o;
    if constexpr (WRITE_BF16) outb[(size_t)row * D_MODEL + c] = (bf16_t)o;
  }
}

// ---------------------------------------------------------------------------
extern "C" void kernel_launch(void* const* d_in, const int* in_sizes, int n_in,
                              void* d_out, int out_size, void* d_ws, size_t ws_size,
                              hipStream_t stream) {
  (void)in_sizes; (void)n_in; (void)out_size; (void)ws_size;
  const float* x = (const float*)d_in[0];
  const float* Wq = (const float*)d_in[1];
  const float* Wk = (const float*)d_in[2];
  const float* Wv = (const float*)d_in[3];
  const float* Wo = (const float*)d_in[4];
  const float* W1 = (const float*)d_in[5];
  const float* b1 = (const float*)d_in[6];
  const float* W2 = (const float*)d_in[7];
  const float* b2 = (const float*)d_in[8];
  const float* g1 = (const float*)d_in[9];
  const float* be1 = (const float*)d_in[10];
  const float* g2 = (const float*)d_in[11];
  const float* be2 = (const float*)d_in[12];
  float* out = (float*)d_out;

  char* wptr = (char*)d_ws;
  auto alloc = [&](size_t bytes) {
    char* p = wptr;
    wptr += (bytes + 255) & ~(size_t)255;
    return p;
  };
  const size_t TOKD = (size_t)NTOK * D_MODEL;
  bf16_t* xb = (bf16_t*)alloc(TOKD * 2);
  bf16_t* wqb = (bf16_t*)alloc((size_t)D_MODEL * D_MODEL * 2);
  bf16_t* wkb = (bf16_t*)alloc((size_t)D_MODEL * D_MODEL * 2);
  bf16_t* wvb = (bf16_t*)alloc((size_t)D_MODEL * D_MODEL * 2);
  bf16_t* wob = (bf16_t*)alloc((size_t)D_MODEL * D_MODEL * 2);
  bf16_t* w1b = (bf16_t*)alloc((size_t)D_FF * D_MODEL * 2);
  bf16_t* w2b = (bf16_t*)alloc((size_t)D_FF * D_MODEL * 2);
  bf16_t* qbuf = (bf16_t*)alloc(TOKD * 2);
  bf16_t* kbuf = (bf16_t*)alloc(TOKD * 2);
  bf16_t* vtb = (bf16_t*)alloc(TOKD * 2);
  bf16_t* ctxb = (bf16_t*)alloc(TOKD * 2);
  float* attnf = (float*)alloc(TOKD * 4);
  float* hf = (float*)alloc(TOKD * 4);
  bf16_t* hb = (bf16_t*)alloc(TOKD * 2);
  bf16_t* f1b = (bf16_t*)alloc((size_t)NTOK * D_FF * 2);
  float* f2f = (float*)alloc(TOKD * 4);

  const dim3 blk(32, 8);

  // bf16 conversions
  cvt_kernel<<<(int)(TOKD / 4 / 256), 256, 0, stream>>>(x, xb, (int)(TOKD / 4));
  int wn4 = D_MODEL * D_MODEL / 4;
  cvt_kernel<<<(wn4 + 255) / 256, 256, 0, stream>>>(Wq, wqb, wn4);
  cvt_kernel<<<(wn4 + 255) / 256, 256, 0, stream>>>(Wk, wkb, wn4);
  cvt_kernel<<<(wn4 + 255) / 256, 256, 0, stream>>>(Wv, wvb, wn4);
  cvt_kernel<<<(wn4 + 255) / 256, 256, 0, stream>>>(Wo, wob, wn4);
  int fn4 = D_FF * D_MODEL / 4;
  cvt_kernel<<<(fn4 + 255) / 256, 256, 0, stream>>>(W1, w1b, fn4);
  cvt_kernel<<<(fn4 + 255) / 256, 256, 0, stream>>>(W2, w2b, fn4);

  // QKV projections. Q carries the softmax scale 1/sqrt(64) and the base-2
  // conversion factor log2(e) folded into its epilogue (exact-enough in bf16).
  const float QSCALE = 0.125f * 1.44269504088896340736f;
  gemm_kernel<false, false, 0, bf16_t>
      <<<dim3(D_MODEL / 128, NTOK / 128), blk, 0, stream>>>(xb, wqb, nullptr, qbuf,
                                                            NTOK, D_MODEL, D_MODEL,
                                                            QSCALE);
  gemm_kernel<false, false, 0, bf16_t>
      <<<dim3(D_MODEL / 128, NTOK / 128), blk, 0, stream>>>(xb, wkb, nullptr, kbuf,
                                                            NTOK, D_MODEL, D_MODEL,
                                                            1.0f);
  gemm_kernel<false, false, 1, bf16_t>
      <<<dim3(D_MODEL / 128, NTOK / 128), blk, 0, stream>>>(xb, wvb, nullptr, vtb,
                                                            NTOK, D_MODEL, D_MODEL,
                                                            1.0f);

  // flash attention -> head-concat context (bf16)
  attn_kernel<<<dim3(SEQ / 128, BATCH * NUM_HEADS), blk, 0, stream>>>(qbuf, kbuf, vtb,
                                                                      ctxb);
  // output projection
  gemm_kernel<false, false, 0, float>
      <<<dim3(D_MODEL / 128, NTOK / 128), blk, 0, stream>>>(ctxb, wob, nullptr, attnf,
                                                            NTOK, D_MODEL, D_MODEL,
                                                            1.0f);
  // h = LN(x + attn), also bf16 copy for FFN
  add_ln_kernel<true><<<NTOK, 256, 0, stream>>>(x, attnf, g1, be1, hf, hb);
  // FFN
  gemm_kernel<true, true, 0, bf16_t>
      <<<dim3(D_FF / 128, NTOK / 128), blk, 0, stream>>>(hb, w1b, b1, f1b, NTOK, D_FF,
                                                         D_MODEL, 1.0f);
  gemm_kernel<false, true, 0, float>
      <<<dim3(D_MODEL / 128, NTOK / 128), blk, 0, stream>>>(f1b, w2b, b2, f2f, NTOK,
                                                            D_MODEL, D_FF, 1.0f);
  // out = LN(h + ffn)
  add_ln_kernel<false><<<NTOK, 256, 0, stream>>>(hf, f2f, g2, be2, out, nullptr);
}
